// Seq2Seq_45286135169674
// MI455X (gfx1250) — matrix operand order repaired and checked by reading, not compiled, not run on hardware
//
#include <hip/hip_runtime.h>

typedef __bf16 bf16_t;
typedef __attribute__((ext_vector_type(16))) __bf16 v16bf;
typedef __attribute__((ext_vector_type(8)))  __bf16 v8bf;
typedef __attribute__((ext_vector_type(8)))  float  v8f;

#define DEV_INLINE __device__ __forceinline__

static constexpr int B_SZ  = 64;
static constexpr int S_LEN = 256;
static constexpr int I_DIM = 64;
static constexpr int H_DIM = 1024;
static constexpr int O_DIM = 64;
static constexpr int T_OUT = 128;
static constexpr int NBLK  = H_DIM / 16;   // 64 persistent blocks

DEV_INLINE float fast_rcp(float x) { return __builtin_amdgcn_rcpf(x); }
DEV_INLINE float sigmoid_f(float x) { return fast_rcp(1.0f + __expf(-x)); }
DEV_INLINE float tanh_f(float x) {
  float e = __expf(2.0f * x);
  return (e - 1.0f) * fast_rcp(e + 1.0f);
}

DEV_INLINE v8f wmma_bf16(v16bf a, v16bf b, v8f c) {
  // (neg_a, A, neg_b, B, c_mod, C, reuse_a, reuse_b)
  return __builtin_amdgcn_wmma_f32_16x16x32_bf16(false, a, false, b, (short)0, c,
                                                 false, false);
}

DEV_INLINE v16bf make_a(v8bf lo, v8bf hi) {
  v16bf a;
#pragma unroll
  for (int q = 0; q < 8; ++q) { a[q] = lo[q]; a[q + 8] = hi[q]; }
  return a;
}

// Low 32 bits of a flat LDS address == wave-relative LDS byte offset.
DEV_INLINE unsigned lds_lo32(const void* p) { return (unsigned)(uintptr_t)p; }

// CDNA5 async global->LDS copy (bypasses VGPRs, tracked by ASYNCcnt).
DEV_INLINE void async_ld_b64(unsigned lds_addr, const void* gaddr) {
  asm volatile("global_load_async_to_lds_b64 %0, %1, off"
               :: "v"(lds_addr), "v"((unsigned long long)(uintptr_t)gaddr)
               : "memory");
}
DEV_INLINE void wait_async_le2() { asm volatile("s_wait_asynccnt 0x2" ::: "memory"); }
DEV_INLINE void wait_async_le0() { asm volatile("s_wait_asynccnt 0x0" ::: "memory"); }

// ---------------------------------------------------------------------------
// Software grid barrier (agent scope).  64 co-resident blocks.
// ---------------------------------------------------------------------------
DEV_INLINE void gsync(unsigned* cnt, unsigned* gen) {
  __builtin_amdgcn_fence(__ATOMIC_RELEASE, "agent");
  __syncthreads();
  if (threadIdx.x == 0) {
    unsigned g = __hip_atomic_load(gen, __ATOMIC_RELAXED, __HIP_MEMORY_SCOPE_AGENT);
    unsigned prev = __hip_atomic_fetch_add(cnt, 1u, __ATOMIC_ACQ_REL,
                                           __HIP_MEMORY_SCOPE_AGENT);
    if (prev == gridDim.x - 1) {
      __hip_atomic_store(cnt, 0u, __ATOMIC_RELAXED, __HIP_MEMORY_SCOPE_AGENT);
      __hip_atomic_fetch_add(gen, 1u, __ATOMIC_RELEASE, __HIP_MEMORY_SCOPE_AGENT);
    } else {
      while (__hip_atomic_load(gen, __ATOMIC_RELAXED, __HIP_MEMORY_SCOPE_AGENT) == g)
        __builtin_amdgcn_s_sleep(1);
    }
  }
  __syncthreads();
  __builtin_amdgcn_fence(__ATOMIC_ACQUIRE, "agent");
}

// ---------------------------------------------------------------------------
// One GEMM phase: acc += A(B x K) @ W-rows^T, K-chunks of 32 double-buffered
// through LDS via async copies.  Block covers 16 hidden units x 4 gates.
// ---------------------------------------------------------------------------
DEV_INLINE void gemm_phase(v8f& acc,
                           const bf16_t* __restrict__ Asrc, int lda,
                           const bf16_t* __restrict__ Wsrc, int ldw, int K,
                           int hbase, bf16_t* a_sh, bf16_t* b_sh) {
  const int tid   = threadIdx.x;
  const int lane  = tid & 31;
  const int wave  = tid >> 5;
  const int gw    = wave & 3;
  const int mw    = wave >> 2;
  const int lh    = lane & 15;
  const int hhalf = lane >> 4;

  // staging role: 512 threads cover 64 rows x 32 K (8 bytes each)
  const int srow = tid >> 3;          // 0..63
  const int skq  = (tid & 7) * 4;     // 0,4,...,28
  const bf16_t* ag = Asrc + (size_t)srow * lda + skq;
  const int wrow   = (srow >> 4) * H_DIM + hbase + (srow & 15);
  const bf16_t* bg = Wsrc + (size_t)wrow * ldw + skq;
  const unsigned aoff = lds_lo32(a_sh + srow * 32 + skq);
  const unsigned boff = lds_lo32(b_sh + srow * 32 + skq);

  const int nk = K >> 5;
  // prologue: chunk 0 -> buffer 0
  async_ld_b64(aoff, ag);
  async_ld_b64(boff, bg);

  for (int i = 0; i < nk; ++i) {
    const int cur = i & 1;
    if (i + 1 < nk) {
      const int nb = (i + 1) & 1;
      const int k0 = (i + 1) << 5;
      async_ld_b64(aoff + nb * 4096, ag + k0);
      async_ld_b64(boff + nb * 4096, bg + k0);
      wait_async_le2();   // only the 2 just-issued may remain outstanding
    } else {
      wait_async_le0();
    }
    __syncthreads();

    const bf16_t* ab = a_sh + cur * 2048 + (mw * 16 + lh) * 32 + hhalf * 16;
    v8bf alo = *(const v8bf*)ab;
    v8bf ahi = *(const v8bf*)(ab + 8);
    const bf16_t* bb = b_sh + cur * 2048 + (gw * 16 + lh) * 32 + hhalf * 16;
    v16bf b = *(const v16bf*)bb;
    acc = wmma_bf16(make_a(alo, ahi), b, acc);

    __syncthreads();     // protect WAR on the buffer we refill next iteration
  }
}

// ---------------------------------------------------------------------------
// Fused LSTM cell step for this block's 16 hidden units.
// ---------------------------------------------------------------------------
DEV_INLINE void cell_step(const bf16_t* __restrict__ X, int ldx, int Kx,
                          const bf16_t* __restrict__ Hprev,
                          const bf16_t* __restrict__ Wih,
                          const bf16_t* __restrict__ Whh,
                          const float* __restrict__ bih,
                          const float* __restrict__ bhh,
                          float* __restrict__ Cst,
                          bf16_t* __restrict__ Hnext,
                          bf16_t* a_sh, bf16_t* b_sh, float* gl) {
  const int tid   = threadIdx.x;
  const int lane  = tid & 31;
  const int wave  = tid >> 5;
  const int gw    = wave & 3;
  const int mw    = wave >> 2;
  const int lh    = lane & 15;
  const int hhalf = lane >> 4;
  const int hbase = blockIdx.x * 16;

  v8f acc = {};
  gemm_phase(acc, X, ldx, Wih, Kx, Kx, hbase, a_sh, b_sh);
  gemm_phase(acc, Hprev, H_DIM, Whh, H_DIM, H_DIM, hbase, a_sh, b_sh);

  // gates -> LDS: gl[gate][batch][16]
#pragma unroll
  for (int r = 0; r < 8; ++r)
    gl[(gw * B_SZ + mw * 16 + hhalf * 8 + r) * 16 + lh] = acc[r];
  __syncthreads();

  for (int e = tid; e < B_SZ * 16; e += 512) {
    int bb = e >> 4;
    int j  = e & 15;
    int hj = hbase + j;
    float gi = gl[(0 * B_SZ + bb) * 16 + j] + bih[hj]             + bhh[hj];
    float gf = gl[(1 * B_SZ + bb) * 16 + j] + bih[H_DIM + hj]     + bhh[H_DIM + hj];
    float gg = gl[(2 * B_SZ + bb) * 16 + j] + bih[2 * H_DIM + hj] + bhh[2 * H_DIM + hj];
    float go = gl[(3 * B_SZ + bb) * 16 + j] + bih[3 * H_DIM + hj] + bhh[3 * H_DIM + hj];
    float iv = sigmoid_f(gi);
    float fv = sigmoid_f(gf);
    float gv = tanh_f(gg);
    float ov = sigmoid_f(go);
    size_t idx = (size_t)bb * H_DIM + hj;
    float c = Cst[idx];
    c = fv * c + iv * gv;
    Cst[idx]   = c;
    Hnext[idx] = (bf16_t)(ov * tanh_f(c));
  }
  __syncthreads();
}

// ---------------------------------------------------------------------------
// Decoder head: out(B x 64) = A(B x H) @ W^T + bias, single block, 16 waves.
// ---------------------------------------------------------------------------
DEV_INLINE void head_gemm_dev(const bf16_t* __restrict__ A,
                              const bf16_t* __restrict__ W,
                              const float* __restrict__ bias,
                              float* __restrict__ out_f32, int out_ld,
                              bf16_t* __restrict__ out_bf16) {
  const int tid   = threadIdx.x;
  const int lane  = tid & 31;
  const int wave  = tid >> 5;
  const int nw    = wave & 3;
  const int mw    = wave >> 2;
  const int lh    = lane & 15;
  const int hhalf = lane >> 4;

  const int m    = mw * 16 + lh;
  const int nrow = nw * 16 + lh;

  v8f acc = {};
  const bf16_t* arow = A + (size_t)m * H_DIM    + hhalf * 16;
  const bf16_t* brow = W + (size_t)nrow * H_DIM + hhalf * 16;
#pragma unroll 4
  for (int k0 = 0; k0 < H_DIM; k0 += 32) {
    v8bf alo = *(const v8bf*)(arow + k0);
    v8bf ahi = *(const v8bf*)(arow + k0 + 8);
    v16bf b  = *(const v16bf*)(brow + k0);
    acc = wmma_bf16(make_a(alo, ahi), b, acc);
  }
#pragma unroll
  for (int r = 0; r < 8; ++r) {
    int mm = mw * 16 + hhalf * 8 + r;
    int nn = nw * 16 + lh;
    float v = acc[r] + bias[nn];
    if (out_f32)  out_f32[(size_t)mm * out_ld + nn] = v;
    if (out_bf16) out_bf16[(size_t)mm * O_DIM + nn] = (bf16_t)v;
  }
}

// ---------------------------------------------------------------------------
struct SeqParams {
  const bf16_t *src;
  const bf16_t *eWih0, *eWhh0, *eWih1, *eWhh1;
  const bf16_t *dWih0, *dWhh0, *dWih1, *dWhh1;
  const bf16_t *Wout, *Win;
  const float  *e_bih0, *e_bhh0, *e_bih1, *e_bhh1;
  const float  *d_bih0, *d_bhh0, *d_bih1, *d_bhh1;
  const float  *bout, *b_in;
  bf16_t *h0a, *h0b, *h1a, *h1b;
  float  *c0, *c1;
  bf16_t *xbuf;
  float  *preds;
  unsigned *cnt, *gen;
};

__global__ __launch_bounds__(512, 1) void seq2seq_persistent(SeqParams p) {
  __shared__ bf16_t a_sh[2 * 2048];            // 8 KB (double-buffered 64x32)
  __shared__ bf16_t b_sh[2 * 2048];            // 8 KB
  __shared__ float  gl[4 * B_SZ * 16];         // 16 KB gate exchange

  bf16_t *h0r = p.h0a, *h0w = p.h0b, *h1r = p.h1a, *h1w = p.h1b;

  // -------- encoder: 256 steps x 2 layers --------
#pragma unroll 1
  for (int t = 0; t < S_LEN; ++t) {
    cell_step(p.src + (size_t)t * I_DIM, S_LEN * I_DIM, I_DIM,
              h0r, p.eWih0, p.eWhh0, p.e_bih0, p.e_bhh0, p.c0, h0w,
              a_sh, b_sh, gl);
    gsync(p.cnt, p.gen);
    { bf16_t* tmp = h0r; h0r = h0w; h0w = tmp; }
    cell_step(h0r, H_DIM, H_DIM,
              h1r, p.eWih1, p.eWhh1, p.e_bih1, p.e_bhh1, p.c1, h1w,
              a_sh, b_sh, gl);
    gsync(p.cnt, p.gen);
    { bf16_t* tmp = h1r; h1r = h1w; h1w = tmp; }
  }

  // -------- decoder: 128 steps --------
#pragma unroll 1
  for (int t = 0; t < T_OUT; ++t) {
    const bf16_t* xp = (t == 0) ? (p.src + (size_t)(S_LEN - 1) * I_DIM) : p.xbuf;
    const int     xl = (t == 0) ? (S_LEN * I_DIM) : I_DIM;
    cell_step(xp, xl, I_DIM,
              h0r, p.dWih0, p.dWhh0, p.d_bih0, p.d_bhh0, p.c0, h0w,
              a_sh, b_sh, gl);
    gsync(p.cnt, p.gen);
    { bf16_t* tmp = h0r; h0r = h0w; h0w = tmp; }
    cell_step(h0r, H_DIM, H_DIM,
              h1r, p.dWih1, p.dWhh1, p.d_bih1, p.d_bhh1, p.c1, h1w,
              a_sh, b_sh, gl);
    gsync(p.cnt, p.gen);
    { bf16_t* tmp = h1r; h1r = h1w; h1w = tmp; }

    if (blockIdx.x == 0) {
      head_gemm_dev(h1r, p.Wout, p.bout,
                    p.preds + (size_t)t * O_DIM, T_OUT * O_DIM, (bf16_t*)nullptr);
    } else if (blockIdx.x == 1) {
      head_gemm_dev(h1r, p.Win, p.b_in, (float*)nullptr, 0, p.xbuf);
    }
    gsync(p.cnt, p.gen);
  }
}

// ---------------------------------------------------------------------------
__global__ void cvt_f32_bf16_kernel(const float* __restrict__ in,
                                    bf16_t* __restrict__ out, int n) {
  int i = blockIdx.x * 256 + threadIdx.x;
  if (i < n) out[i] = (bf16_t)in[i];
}

__global__ void zero_u32_kernel(unsigned* __restrict__ p, int n) {
  int i = blockIdx.x * 256 + threadIdx.x;
  if (i < n) p[i] = 0u;
}

// ---------------------------------------------------------------------------
extern "C" void kernel_launch(void* const* d_in, const int* in_sizes, int n_in,
                              void* d_out, int out_size, void* d_ws, size_t ws_size,
                              hipStream_t stream) {
  (void)in_sizes; (void)n_in; (void)out_size; (void)ws_size;

  const float* src    = (const float*)d_in[0];
  const float* eWih0  = (const float*)d_in[1];
  const float* eWhh0  = (const float*)d_in[2];
  const float* e_bih0 = (const float*)d_in[3];
  const float* e_bhh0 = (const float*)d_in[4];
  const float* eWih1  = (const float*)d_in[5];
  const float* eWhh1  = (const float*)d_in[6];
  const float* e_bih1 = (const float*)d_in[7];
  const float* e_bhh1 = (const float*)d_in[8];
  const float* dWih0  = (const float*)d_in[9];
  const float* dWhh0  = (const float*)d_in[10];
  const float* d_bih0 = (const float*)d_in[11];
  const float* d_bhh0 = (const float*)d_in[12];
  const float* dWih1  = (const float*)d_in[13];
  const float* dWhh1  = (const float*)d_in[14];
  const float* d_bih1 = (const float*)d_in[15];
  const float* d_bhh1 = (const float*)d_in[16];
  const float* Wout   = (const float*)d_in[17];
  const float* bout   = (const float*)d_in[18];
  const float* Win    = (const float*)d_in[19];
  const float* b_in   = (const float*)d_in[20];

  char* ws = (char*)d_ws;
  size_t off = 0;
  auto alloc = [&](size_t bytes) -> char* {
    off = (off + 255) & ~(size_t)255;
    char* p = ws + off;
    off += bytes;
    return p;
  };

  const int G4H = 4 * H_DIM;
  SeqParams p;
  bf16_t* src_bf = (bf16_t*)alloc((size_t)B_SZ * S_LEN * I_DIM * 2);
  bf16_t* w_eWih0 = (bf16_t*)alloc((size_t)G4H * I_DIM * 2);
  bf16_t* w_eWhh0 = (bf16_t*)alloc((size_t)G4H * H_DIM * 2);
  bf16_t* w_eWih1 = (bf16_t*)alloc((size_t)G4H * H_DIM * 2);
  bf16_t* w_eWhh1 = (bf16_t*)alloc((size_t)G4H * H_DIM * 2);
  bf16_t* w_dWih0 = (bf16_t*)alloc((size_t)G4H * I_DIM * 2);
  bf16_t* w_dWhh0 = (bf16_t*)alloc((size_t)G4H * H_DIM * 2);
  bf16_t* w_dWih1 = (bf16_t*)alloc((size_t)G4H * H_DIM * 2);
  bf16_t* w_dWhh1 = (bf16_t*)alloc((size_t)G4H * H_DIM * 2);
  bf16_t* w_Wout  = (bf16_t*)alloc((size_t)O_DIM * H_DIM * 2);
  bf16_t* w_Win   = (bf16_t*)alloc((size_t)I_DIM * H_DIM * 2);
  p.h0a  = (bf16_t*)alloc((size_t)B_SZ * H_DIM * 2);
  p.h0b  = (bf16_t*)alloc((size_t)B_SZ * H_DIM * 2);
  p.h1a  = (bf16_t*)alloc((size_t)B_SZ * H_DIM * 2);
  p.h1b  = (bf16_t*)alloc((size_t)B_SZ * H_DIM * 2);
  p.c0   = (float*)alloc((size_t)B_SZ * H_DIM * 4);
  p.c1   = (float*)alloc((size_t)B_SZ * H_DIM * 4);
  p.xbuf = (bf16_t*)alloc((size_t)B_SZ * I_DIM * 2);
  unsigned* sync_mem = (unsigned*)alloc(256);

  p.src = src_bf;
  p.eWih0 = w_eWih0; p.eWhh0 = w_eWhh0; p.eWih1 = w_eWih1; p.eWhh1 = w_eWhh1;
  p.dWih0 = w_dWih0; p.dWhh0 = w_dWhh0; p.dWih1 = w_dWih1; p.dWhh1 = w_dWhh1;
  p.Wout = w_Wout; p.Win = w_Win;
  p.e_bih0 = e_bih0; p.e_bhh0 = e_bhh0; p.e_bih1 = e_bih1; p.e_bhh1 = e_bhh1;
  p.d_bih0 = d_bih0; p.d_bhh0 = d_bhh0; p.d_bih1 = d_bih1; p.d_bhh1 = d_bhh1;
  p.bout = bout; p.b_in = b_in;
  p.preds = (float*)d_out;
  p.cnt = sync_mem;
  p.gen = sync_mem + 1;

  auto cvt = [&](const float* in, bf16_t* dst, int n) {
    cvt_f32_bf16_kernel<<<(n + 255) / 256, 256, 0, stream>>>(in, dst, n);
  };
  auto zero = [&](void* q, size_t bytes) {
    int n = (int)(bytes / 4);
    zero_u32_kernel<<<(n + 255) / 256, 256, 0, stream>>>((unsigned*)q, n);
  };

  cvt(src,   src_bf,  B_SZ * S_LEN * I_DIM);
  cvt(eWih0, w_eWih0, G4H * I_DIM);
  cvt(eWhh0, w_eWhh0, G4H * H_DIM);
  cvt(eWih1, w_eWih1, G4H * H_DIM);
  cvt(eWhh1, w_eWhh1, G4H * H_DIM);
  cvt(dWih0, w_dWih0, G4H * I_DIM);
  cvt(dWhh0, w_dWhh0, G4H * H_DIM);
  cvt(dWih1, w_dWih1, G4H * H_DIM);
  cvt(dWhh1, w_dWhh1, G4H * H_DIM);
  cvt(Wout,  w_Wout,  O_DIM * H_DIM);
  cvt(Win,   w_Win,   I_DIM * H_DIM);

  zero(p.h0a, (size_t)B_SZ * H_DIM * 2);
  zero(p.h1a, (size_t)B_SZ * H_DIM * 2);
  zero(p.c0,  (size_t)B_SZ * H_DIM * 4);
  zero(p.c1,  (size_t)B_SZ * H_DIM * 4);
  zero(sync_mem, 256);

  seq2seq_persistent<<<NBLK, 512, 0, stream>>>(p);
}
